// VoxelSetAbstractionRange_81982335746526
// MI455X (gfx1250) — compile-verified, matrix operand order checked by reading, not run on hardware
//
#include <hip/hip_runtime.h>
#include <hip/hip_bf16.h>

typedef __attribute__((ext_vector_type(16))) _Float16 v16h;
typedef __attribute__((ext_vector_type(8)))  _Float16 v8h;
typedef __attribute__((ext_vector_type(8)))  float    v8f;

#define NPTS     16384
#define BATCH    2
#define KTOT     2048
#define TOTAL_KP 4096
#define HBEV     200
#define WBEV     176
#define CBEV     256
#define CIN      288
#define COUT     128
#define EPSF     1e-5f

// ---------------------------------------------------------------------------
// 1) Furthest-point sampling, one 32-wave workgroup per (batch, band).
//    Per-thread min-distances live in registers; LDS tree argmax per step.
// ---------------------------------------------------------------------------
__global__ __launch_bounds__(1024) void fps_kernel(const float* __restrict__ points,
                                                   const float* __restrict__ rng,
                                                   int* __restrict__ selIdx) {
  const int blk  = blockIdx.x;            // 0..5
  const int b    = blk / 3, band = blk % 3;
  const int K    = (band == 0) ? 1024 : 512;
  const int bandOff = (band == 0) ? 0 : (band == 1 ? 1024 : 1536);
  int* sel = selIdx + b * KTOT + bandOff;
  const int t = threadIdx.x;

  __shared__ float sVal[1024];
  __shared__ int   sIdx[1024];
  __shared__ float sLast[3];

  float dist[16];
  int firstLocal = NPTS;
#pragma unroll
  for (int j = 0; j < 16; ++j) {
    int p = j * 1024 + t;
    float r = rng[b * NPTS + p];
    bool shortm = (r > 0.0f) && (r < 25.0f);
    bool longm  = (r > 45.0f);
    bool valid  = (band == 0) ? shortm : ((band == 2) ? longm : (!shortm && !longm));
    dist[j] = valid ? 1e10f : -1.0f;
    if (valid && p < firstLocal) firstLocal = p;
  }
  sIdx[t] = firstLocal;
  __syncthreads();
  for (int off = 512; off > 0; off >>= 1) {
    if (t < off) { int o = sIdx[t + off]; if (o < sIdx[t]) sIdx[t] = o; }
    __syncthreads();
  }
  int first = sIdx[0];
  if (first >= NPTS) first = 0;
  if (t == 0) {
    sel[0] = first;
    const float* pp = points + (size_t)(b * NPTS + first) * 5;
    sLast[0] = pp[1]; sLast[1] = pp[2]; sLast[2] = pp[3];
  }
  __syncthreads();

  for (int s = 1; s < K; ++s) {
    float lx = sLast[0], ly = sLast[1], lz = sLast[2];
    float bv = -2.0f; int bi = 0x7fffffff;
#pragma unroll
    for (int j = 0; j < 16; ++j) {
      int p = j * 1024 + t;
      if (dist[j] >= 0.0f) {
        const float* pp = points + (size_t)(b * NPTS + p) * 5;
        float dx = pp[1] - lx, dy = pp[2] - ly, dz = pp[3] - lz;
        float d = dx * dx + dy * dy + dz * dz;
        dist[j] = fminf(dist[j], d);
      }
      float v = dist[j];
      if (v > bv || (v == bv && p < bi)) { bv = v; bi = p; }
    }
    sVal[t] = bv; sIdx[t] = bi;
    __syncthreads();
    for (int off = 512; off > 0; off >>= 1) {
      if (t < off) {
        float v2 = sVal[t + off]; int i2 = sIdx[t + off];
        if (v2 > sVal[t] || (v2 == sVal[t] && i2 < sIdx[t])) { sVal[t] = v2; sIdx[t] = i2; }
      }
      __syncthreads();
    }
    if (t == 0) {
      int nxt = sIdx[0];
      sel[s] = nxt;
      const float* pp = points + (size_t)(b * NPTS + nxt) * 5;
      sLast[0] = pp[1]; sLast[1] = pp[2]; sLast[2] = pp[3];
    }
    __syncthreads();
  }
}

// ---------------------------------------------------------------------------
// 2) Gather keypoint coords; emit point_coords output section.
// ---------------------------------------------------------------------------
__global__ void gather_kernel(const float* __restrict__ points,
                              const int* __restrict__ selIdx,
                              float* __restrict__ kp,
                              float* __restrict__ outCoords) {
  int j = blockIdx.x * blockDim.x + threadIdx.x;
  if (j >= TOTAL_KP) return;
  int b = j >> 11;
  int idx = selIdx[j];
  const float* pp = points + (size_t)(b * NPTS + idx) * 5;
  float x = pp[1], y = pp[2], z = pp[3];
  kp[j * 3 + 0] = x; kp[j * 3 + 1] = y; kp[j * 3 + 2] = z;
  outCoords[j * 4 + 0] = (float)b;
  outCoords[j * 4 + 1] = x;
  outCoords[j * 4 + 2] = y;
  outCoords[j * 4 + 3] = z;
}

// ---------------------------------------------------------------------------
// 3) Bilinear BEV sampling: block per keypoint, thread per channel (coalesced
//    along W stride; 72MB map is L2-resident on 192MB L2).
// ---------------------------------------------------------------------------
__global__ __launch_bounds__(256) void bilinear_kernel(const float* __restrict__ sf,
                                                       const float* __restrict__ kp,
                                                       float* __restrict__ outFeats) {
  int j = blockIdx.x;
  int c = threadIdx.x;
  int b = j >> 11;
  float x = kp[j * 3 + 0], y = kp[j * 3 + 1];
  float xi = (x - 0.0f) / 0.05f / 8.0f;
  float yi = (y - (-40.0f)) / 0.05f / 8.0f;
  int x0 = (int)floorf(xi), x1 = x0 + 1;
  int y0 = (int)floorf(yi), y1 = y0 + 1;
  x0 = min(max(x0, 0), WBEV - 1); x1 = min(max(x1, 0), WBEV - 1);
  y0 = min(max(y0, 0), HBEV - 1); y1 = min(max(y1, 0), HBEV - 1);
  float wa = ((float)x1 - xi) * ((float)y1 - yi);
  float wb = ((float)x1 - xi) * (yi - (float)y0);
  float wc = (xi - (float)x0) * ((float)y1 - yi);
  float wd = (xi - (float)x0) * (yi - (float)y0);
  const float* base = sf + ((size_t)b * CBEV + c) * (HBEV * WBEV);
  float Ia = base[y0 * WBEV + x0];
  float Ib = base[y1 * WBEV + x0];
  float Ic = base[y0 * WBEV + x1];
  float Id = base[y1 * WBEV + x1];
  outFeats[(size_t)j * CIN + c] = Ia * wa + Ib * wb + Ic * wc + Id * wd;
}

// ---------------------------------------------------------------------------
// 4) SA module: one wave32 per keypoint. Ballot/popc ordered compaction gives
//    "first ns valid indices ascending"; lane = (branch, sample) for the MLP;
//    shfl_xor max-pool over each 16-lane group.
// ---------------------------------------------------------------------------
__global__ __launch_bounds__(256) void sa_kernel(const float* __restrict__ points,
                                                 const float* __restrict__ kp,
                                                 const float* __restrict__ w0,
                                                 const float* __restrict__ bn0,
                                                 const float* __restrict__ w1,
                                                 const float* __restrict__ bn1,
                                                 float* __restrict__ outFeats) {
  const int w    = threadIdx.x >> 5;
  const int lane = threadIdx.x & 31;
  const int j    = blockIdx.x * 8 + w;
  const int b    = j >> 11;
  __shared__ int nIdx[8][2][16];

  float kx = kp[j * 3 + 0], ky = kp[j * 3 + 1], kz = kp[j * 3 + 2];
  int cnt0 = 0, cnt1 = 0;
  for (int c = 0; c < NPTS / 32; ++c) {
    int p = c * 32 + lane;
    const float* pp = points + (size_t)(b * NPTS + p) * 5;
    float dx = pp[1] - kx, dy = pp[2] - ky, dz = pp[3] - kz;
    float d2 = dx * dx + dy * dy + dz * dz;
    unsigned int m1 = (unsigned int)__ballot(d2 < 1.0f);
    unsigned int m2 = (unsigned int)__ballot(d2 < 4.0f);
    unsigned int pre = (1u << lane) - 1u;
    if (cnt0 < 16) {
      int pos = cnt0 + __popc(m1 & pre);
      if (((m1 >> lane) & 1u) && pos < 16) nIdx[w][0][pos] = p;
    }
    if (cnt1 < 16) {
      int pos = cnt1 + __popc(m2 & pre);
      if (((m2 >> lane) & 1u) && pos < 16) nIdx[w][1][pos] = p;
    }
    cnt0 += __popc(m1);
    cnt1 += __popc(m2);
    if (cnt0 >= 16 && cnt1 >= 16) break;   // wave-uniform
  }
  __syncthreads();

  const int br = lane >> 4, s = lane & 15;
  const int cnt = br ? cnt1 : cnt0;
  float g0, g1, g2, g3;
  if (cnt == 0) {
    g0 = g1 = g2 = g3 = 0.0f;              // empty group: features zeroed
  } else {
    int sc = (s < cnt) ? s : 0;            // pad with first neighbor
    int gidx = nIdx[w][br][sc];
    const float* pp = points + (size_t)(b * NPTS + gidx) * 5;
    g0 = pp[1] - kx; g1 = pp[2] - ky; g2 = pp[3] - kz; g3 = pp[4];
  }
  float h1[16];
#pragma unroll
  for (int c = 0; c < 16; ++c) {
    float a = g0 * w0[(br * 4 + 0) * 16 + c] + g1 * w0[(br * 4 + 1) * 16 + c]
            + g2 * w0[(br * 4 + 2) * 16 + c] + g3 * w0[(br * 4 + 3) * 16 + c];
    float gm = bn0[(br * 4 + 0) * 16 + c], bt = bn0[(br * 4 + 1) * 16 + c];
    float mn = bn0[(br * 4 + 2) * 16 + c], vr = bn0[(br * 4 + 3) * 16 + c];
    float scl = gm * rsqrtf(vr + EPSF);
    h1[c] = fmaxf((a - mn) * scl + bt, 0.0f);
  }
  float h2[16];
#pragma unroll
  for (int c = 0; c < 16; ++c) {
    float a = 0.0f;
#pragma unroll
    for (int q = 0; q < 16; ++q) a += h1[q] * w1[(br * 16 + q) * 16 + c];
    float gm = bn1[(br * 4 + 0) * 16 + c], bt = bn1[(br * 4 + 1) * 16 + c];
    float mn = bn1[(br * 4 + 2) * 16 + c], vr = bn1[(br * 4 + 3) * 16 + c];
    float scl = gm * rsqrtf(vr + EPSF);
    h2[c] = fmaxf((a - mn) * scl + bt, 0.0f);
  }
#pragma unroll
  for (int off = 8; off >= 1; off >>= 1) {
#pragma unroll
    for (int c = 0; c < 16; ++c)
      h2[c] = fmaxf(h2[c], __shfl_xor(h2[c], off, 32));
  }
  if (s == 0) {
#pragma unroll
    for (int c = 0; c < 16; ++c)
      outFeats[(size_t)j * CIN + CBEV + br * 16 + c] = h2[c];
  }
}

// ---------------------------------------------------------------------------
// 5) f32 -> f16 conversions for the WMMA GEMM (+ W transpose, BN fold).
// ---------------------------------------------------------------------------
__global__ void cvt_feats_kernel(const float* __restrict__ feats,
                                 _Float16* __restrict__ out) {
  int i = blockIdx.x * blockDim.x + threadIdx.x;
  if (i < TOTAL_KP * CIN) out[i] = (_Float16)feats[i];
}

__global__ void cvt_w_kernel(const float* __restrict__ fw, const float* __restrict__ fbn,
                             _Float16* __restrict__ wt, float* __restrict__ bnss) {
  int i = blockIdx.x * blockDim.x + threadIdx.x;
  if (i < COUT * CIN) {
    int n = i / CIN, k = i % CIN;
    wt[n * CIN + k] = (_Float16)fw[k * COUT + n];   // transpose to [N, K]
  }
  if (i < COUT) {
    float sc = fbn[0 * COUT + i] * rsqrtf(fbn[3 * COUT + i] + EPSF);
    bnss[i] = sc;                                   // scale
    bnss[COUT + i] = fbn[1 * COUT + i] - fbn[2 * COUT + i] * sc;  // shift
  }
}

// ---------------------------------------------------------------------------
// 6) Fusion GEMM: one wave per 16x16 tile, K=288 in 9 unrolled 16x16x32 WMMAs.
//    A/B lane loads follow the CDNA5 16-bit striping: lanes 0-15 own
//    K in {kk..kk+7, kk+16..kk+23}; lanes 16-31 the +8 halves. Two b128
//    loads per operand per step. BN scale/shift + ReLU in the epilogue.
// ---------------------------------------------------------------------------
__global__ __launch_bounds__(128) void fusion_wmma_kernel(const _Float16* __restrict__ A,
                                                          const _Float16* __restrict__ Bt,
                                                          const float* __restrict__ bnss,
                                                          float* __restrict__ out) {
  const int wave = blockIdx.x * (blockDim.x >> 5) + (threadIdx.x >> 5);
  const int lane = threadIdx.x & 31;
  const int tileM = (wave >> 3) << 4;   // 256 tiles along M (4096/16)
  const int tileN = (wave & 7) << 4;    // 8 tiles along N (128/16)
  const int hsel  = lane >> 4;
  const int l15   = lane & 15;
  const int kBase = hsel ? 8 : 0;

  const _Float16* aRow = A  + (size_t)(tileM + l15) * CIN + kBase;
  const _Float16* bRow = Bt + (size_t)(tileN + l15) * CIN + kBase;

  v8f acc = {};
#pragma unroll
  for (int kk = 0; kk < CIN / 32; ++kk) {
    v8h a_lo = *(const v8h*)(aRow + kk * 32);
    v8h a_hi = *(const v8h*)(aRow + kk * 32 + 16);
    v8h b_lo = *(const v8h*)(bRow + kk * 32);
    v8h b_hi = *(const v8h*)(bRow + kk * 32 + 16);
    v16h av = __builtin_shufflevector(a_lo, a_hi, 0,1,2,3,4,5,6,7,8,9,10,11,12,13,14,15);
    v16h bv = __builtin_shufflevector(b_lo, b_hi, 0,1,2,3,4,5,6,7,8,9,10,11,12,13,14,15);
    acc = __builtin_amdgcn_wmma_f32_16x16x32_f16(false, av, false, bv,
                                                 (short)0, acc, false, false);
  }

  const int col = tileN + l15;                // C/D layout: lane&15 = N
  const int rowBase = tileM + (hsel ? 8 : 0); // VGPR r -> M = rowBase + r
  float sc = bnss[col];
  float sh = bnss[COUT + col];
#pragma unroll
  for (int r = 0; r < 8; ++r) {
    float v = acc[r] * sc + sh;
    out[(size_t)(rowBase + r) * COUT + col] = fmaxf(v, 0.0f);
  }
}

// ---------------------------------------------------------------------------
// Launch
// ---------------------------------------------------------------------------
extern "C" void kernel_launch(void* const* d_in, const int* in_sizes, int n_in,
                              void* d_out, int out_size, void* d_ws, size_t ws_size,
                              hipStream_t stream) {
  const float* points = (const float*)d_in[0];
  const float* prange = (const float*)d_in[1];
  const float* sf     = (const float*)d_in[2];
  const float* sa_w0  = (const float*)d_in[3];
  const float* sa_bn0 = (const float*)d_in[4];
  const float* sa_w1  = (const float*)d_in[5];
  const float* sa_bn1 = (const float*)d_in[6];
  const float* fw     = (const float*)d_in[7];
  const float* fbn    = (const float*)d_in[8];

  float* out       = (float*)d_out;
  float* outFused  = out;                                  // [4096,128]
  float* outFeats  = out + TOTAL_KP * COUT;                // [4096,288]
  float* outCoords = out + TOTAL_KP * COUT + TOTAL_KP * CIN; // [4096,4]

  char* ws = (char*)d_ws;
  int*      selIdx = (int*)ws;                                   // 16 KB
  float*    kp     = (float*)(ws + 16384);                       // 48 KB
  _Float16* aF16   = (_Float16*)(ws + 65536);                    // 2.25 MB
  _Float16* wtF16  = (_Float16*)(ws + 65536 + 2359296);          // 72 KB
  float*    bnss   = (float*)(ws + 65536 + 2359296 + 73728);     // 1 KB

  hipLaunchKernelGGL(fps_kernel,      dim3(6),    dim3(1024), 0, stream, points, prange, selIdx);
  hipLaunchKernelGGL(gather_kernel,   dim3(16),   dim3(256),  0, stream, points, selIdx, kp, outCoords);
  hipLaunchKernelGGL(bilinear_kernel, dim3(4096), dim3(256),  0, stream, sf, kp, outFeats);
  hipLaunchKernelGGL(sa_kernel,       dim3(512),  dim3(256),  0, stream, points, kp,
                     sa_w0, sa_bn0, sa_w1, sa_bn1, outFeats);
  hipLaunchKernelGGL(cvt_feats_kernel, dim3((TOTAL_KP * CIN) / 256), dim3(256), 0, stream,
                     outFeats, aF16);
  hipLaunchKernelGGL(cvt_w_kernel, dim3((COUT * CIN + 255) / 256), dim3(256), 0, stream,
                     fw, fbn, wtF16, bnss);
  hipLaunchKernelGGL(fusion_wmma_kernel, dim3(512), dim3(128), 0, stream,
                     aF16, wtF16, bnss, outFused);
}
